// EpisodicMemory_86088324481737
// MI455X (gfx1250) — matrix operand order repaired and checked by reading, not compiled, not run on hardware
//
#include <hip/hip_runtime.h>
#include <hip/hip_bf16.h>
#include <math.h>

// MI455X / gfx1250, wave32. bf16 WMMA 16x16x32, f32 accumulate.

typedef __attribute__((ext_vector_type(16))) __bf16 v16bf;
typedef __attribute__((ext_vector_type(8)))  float  v8f;

#define B_DIM 128
#define S_DIM 256
#define H_DIM 512

// ---- WMMA fragment index helpers (CDNA5 ISA §7.12.2, wave32) ----
// A (16-bit, 16x32): lane holds row M = lane&15; half g = lane>>4.
// pair ep (vgpr), p in {0,1}: K = a_pair_base(ep,g) + p  (pairs are K-consecutive)
__device__ __forceinline__ int a_pair_base(int ep, int g) {
    return ((ep < 4) ? (2 * ep) : (16 + 2 * (ep - 4))) + 8 * g;
}
// B (16-bit, 32x16): lane holds col N = lane&15; K = 8g + e (e<8), 16 + 8g + (e-8) (e>=8)
__device__ __forceinline__ int b_kidx(int e, int g) {
    return ((e < 8) ? e : (16 + (e - 8))) + 8 * g;
}
// C/D (f32 16x16): lane holds col N = lane&15; vgpr i holds row M = i + 8g

union AFrag { v16bf v; unsigned int u[8]; };

// ---------------------------------------------------------------------
// init: zero logits accumulator, C (f32 master), C bf16 buffer 0
__global__ void init_kernel(float* __restrict__ logits,
                            float* __restrict__ Cf,
                            __bf16* __restrict__ Cbf0) {
    int tid = blockIdx.x * blockDim.x + threadIdx.x;   // 65536 threads
    if (tid < B_DIM * S_DIM) logits[tid] = 0.0f;
    Cf[tid] = 0.0f;
    Cbf0[tid] = (__bf16)0.0f;
}

// ---------------------------------------------------------------------
// rowsum(Wr), rowsum(W): the einsum('bsh,hk->bsh') contraction over k
__global__ void rowsum_kernel(const float* __restrict__ Wr,
                              const float* __restrict__ W,
                              float* __restrict__ rsWr,
                              float* __restrict__ rsW) {
    int h = blockIdx.x * blockDim.x + threadIdx.x;     // 512 threads
    if (h >= H_DIM) return;
    float s1 = 0.0f, s2 = 0.0f;
    for (int k = 0; k < H_DIM; ++k) {
        s1 += Wr[(size_t)h * H_DIM + k];
        s2 += W [(size_t)h * H_DIM + k];
    }
    rsWr[h] = s1;
    rsW[h]  = s2;
}

// ---------------------------------------------------------------------
// One-time weight pre-swizzle: f32 (K x 512 row-major) -> bf16 in exact
// B-fragment register order: dst[((ntile*KT + kt)*32 + lane)*16 + e].
// Consumers then load each B fragment as one contiguous 32-byte chunk.
// grid: (K/32, 32), block: 32
__global__ void pack_b_kernel(const float* __restrict__ src,
                              __bf16* __restrict__ dst) {
    const int lane = threadIdx.x & 31;
    const int kt = blockIdx.x;                 // K/32 tiles
    const int nt = blockIdx.y;                 // 32 N tiles
    const int KT = gridDim.x;
    const int lm = lane & 15, g = lane >> 4;
    const int kb = kt * 32;
    __bf16* out = dst + (((size_t)nt * KT + kt) * 32 + lane) * 16;
    #pragma unroll
    for (int e = 0; e < 16; ++e) {
        int k = kb + b_kidx(e, g);
        out[e] = (__bf16)src[(size_t)k * H_DIM + nt * 16 + lm];
    }
}

// ---------------------------------------------------------------------
// Gate GEMM: g_hidden = tanh(z @ z1_w + z1_b); logits += g_hidden @ z2_w
// z built on the fly from an LDS-staged facts tile; 4 straight-line
// sub-loops (one per z block) -> no data-dependent branches.
// grid: (B*S/16, H/64), block: 128 (4 waves, each one 16x16 N-tile)
__global__ void gate_gemm_kernel(const float* __restrict__ facts,
                                 const float* __restrict__ q,
                                 const float* __restrict__ pm,
                                 const __bf16* __restrict__ z1w_pk,
                                 const float* __restrict__ z1b,
                                 const float* __restrict__ z2w,
                                 float* __restrict__ logits) {
    __shared__ float sf[16 * H_DIM];           // facts tile, 32 KB
    __shared__ float sq[H_DIM];                // question row, 2 KB
    __shared__ float sm[H_DIM];                // prevM row, 2 KB

    const int tid  = threadIdx.x;              // 128
    const int lane = tid & 31;
    const int wave = tid >> 5;
    const int r0 = blockIdx.x * 16;            // row tile in [0, B*S)
    const int ntile = blockIdx.y * 4 + wave;   // [0, 32)
    const int lm = lane & 15;
    const int g  = lane >> 4;
    const int bidx = r0 >> 8;                  // 16-row tile lies in one batch

    // coalesced cooperative staging (float4)
    {
        const float4* fsrc = (const float4*)(facts + (size_t)r0 * H_DIM);
        float4* fdst = (float4*)sf;
        #pragma unroll
        for (int i = 0; i < 16; ++i) fdst[tid + 128 * i] = fsrc[tid + 128 * i];
        ((float4*)sq)[tid] = ((const float4*)(q  + (size_t)bidx * H_DIM))[tid];
        ((float4*)sm)[tid] = ((const float4*)(pm + (size_t)bidx * H_DIM))[tid];
    }
    __syncthreads();

    const float* frow = sf + lm * H_DIM;
    v8f acc = {};

    // mode 0: f*q   1: f*m   2: |f-q|   3: |f-m|   (compile-time unrolled)
    // |x| written as fmaxf(a-b, b-a): avoids the packed-fabs bitcast pattern
    // that crashes gfx1250 ISel.
    #pragma unroll
    for (int mode = 0; mode < 4; ++mode) {
        const float* vrow = (mode == 0 || mode == 2) ? sq : sm;
        for (int kb = 0; kb < H_DIM; kb += 32) {
            v16bf afrag;
            #pragma unroll
            for (int ep = 0; ep < 8; ++ep) {
                int col = kb + a_pair_base(ep, g);
                float f0 = frow[col], f1 = frow[col + 1];
                float v0 = vrow[col], v1 = vrow[col + 1];
                float w0, w1;
                if (mode < 2) {
                    w0 = f0 * v0;
                    w1 = f1 * v1;
                } else {
                    w0 = fmaxf(f0 - v0, v0 - f0);
                    w1 = fmaxf(f1 - v1, v1 - f1);
                }
                afrag[2 * ep]     = (__bf16)w0;
                afrag[2 * ep + 1] = (__bf16)w1;
            }
            const int ktg = (mode * H_DIM + kb) >> 5;              // [0, 64)
            const v16bf bfrag =
                *(const v16bf*)(z1w_pk + (((size_t)ntile * 64 + ktg) * 32 + lane) * 16);
            acc = __builtin_amdgcn_wmma_f32_16x16x32_bf16(
                false, afrag, false, bfrag, (short)0, acc, false, false);
        }
    }

    // epilogue: tanh(+bias) * z2w column, shuffle-reduce over the 16 N lanes,
    // single atomic per row per wave. (z2_b is softmax-invariant, omitted)
    const int ncol = ntile * 16 + lm;
    const float zb = z1b[ncol];
    const float zw = z2w[ncol];
    #pragma unroll
    for (int i = 0; i < 8; ++i) {
        float v = tanhf(acc[i] + zb) * zw;
        v += __shfl_xor(v, 1, 32);
        v += __shfl_xor(v, 2, 32);
        v += __shfl_xor(v, 4, 32);
        v += __shfl_xor(v, 8, 32);
        if (lm == 0) atomicAdd(&logits[r0 + i + 8 * g], v);
    }
}

// ---------------------------------------------------------------------
// softmax over S per batch row
__global__ void softmax_kernel(const float* __restrict__ logits,
                               float* __restrict__ G) {
    __shared__ float red[S_DIM];
    const int b = blockIdx.x, t = threadIdx.x;
    float x = logits[b * S_DIM + t];
    red[t] = x;
    __syncthreads();
    for (int off = S_DIM / 2; off > 0; off >>= 1) {
        if (t < off) red[t] = fmaxf(red[t], red[t + off]);
        __syncthreads();
    }
    float mx = red[0];
    __syncthreads();
    float e = expf(x - mx);
    red[t] = e;
    __syncthreads();
    for (int off = S_DIM / 2; off > 0; off >>= 1) {
        if (t < off) red[t] += red[t + off];
        __syncthreads();
    }
    G[b * S_DIM + t] = e / red[0];
}

// ---------------------------------------------------------------------
// One recurrence step s: fused C@Ur and C@U WMMA tiles + GRU-like update.
// grid: (B/16, H/16), block: 32 (one wave per 16x16 output tile)
__global__ void scan_step_kernel(const __bf16* __restrict__ Cin_bf,
                                 __bf16* __restrict__ Cout_bf,
                                 float* __restrict__ Cf,
                                 const float* __restrict__ facts,
                                 const float* __restrict__ G,
                                 const __bf16* __restrict__ Ur_pk,
                                 const __bf16* __restrict__ U_pk,
                                 const float* __restrict__ rsWr,
                                 const float* __restrict__ rsW,
                                 const float* __restrict__ brv,
                                 const float* __restrict__ burv,
                                 const float* __restrict__ bwv,
                                 const float* __restrict__ buv,
                                 int s) {
    const int lane = threadIdx.x & 31;
    const int bt = blockIdx.x;                 // batch tile (8)
    const int ht = blockIdx.y;                 // hidden tile (32)
    const int lm = lane & 15;
    const int g  = lane >> 4;
    const int arow = bt * 16 + lm;             // batch row for A loads

    const unsigned int* cin_u = (const unsigned int*)Cin_bf;
    const size_t crow = ((size_t)arow * H_DIM) >> 1;   // in bf16 pairs

    v8f accr = {}, accu = {};
    for (int kt = 0; kt < 16; ++kt) {          // K = 512, step 32
        const int kb = kt * 32;
        AFrag af;
        #pragma unroll
        for (int ep = 0; ep < 8; ++ep)
            af.u[ep] = cin_u[crow + ((kb + a_pair_base(ep, g)) >> 1)];
        const size_t boff = (((size_t)ht * 16 + kt) * 32 + lane) * 16;
        const v16bf bfr = *(const v16bf*)(Ur_pk + boff);
        const v16bf bfu = *(const v16bf*)(U_pk  + boff);
        accr = __builtin_amdgcn_wmma_f32_16x16x32_bf16(
            false, af.v, false, bfr, (short)0, accr, false, false);
        accu = __builtin_amdgcn_wmma_f32_16x16x32_bf16(
            false, af.v, false, bfu, (short)0, accu, false, false);
    }

    const int h = ht * 16 + lm;
    const float rwr = rsWr[h], rw = rsW[h];
    const float c_br = brv[h], c_bur = burv[h], c_bw = bwv[h], c_bu = buv[h];
    #pragma unroll
    for (int i = 0; i < 8; ++i) {
        const int bidx = bt * 16 + i + 8 * g;
        const float f = facts[((size_t)bidx * S_DIM + s) * H_DIM + h];
        const float pre_r = f * rwr + c_br;    // einsum quirk: rowsum scaling
        const float pre_h = f * rw  + c_bw;
        const float r = 1.0f / (1.0f + expf(-(pre_r + accr[i] + c_bur)));
        const float htil = tanhf(pre_h + r * (accu[i] + c_bu));
        const float gg = G[bidx * S_DIM + s];
        const size_t idx = (size_t)bidx * H_DIM + h;
        const float cold = Cf[idx];
        const float cnew = gg * htil + (1.0f - gg) * cold;
        Cf[idx] = cnew;
        Cout_bf[idx] = (__bf16)cnew;
    }
}

// ---------------------------------------------------------------------
// next_mem = relu(concat[prevM, C, questions] @ nm_w + nm_b)
// 3 straight-line K segments (one per concat block).
// grid: (B/16, H/16), block: 32
__global__ void final_kernel(const float* __restrict__ pm,
                             const float* __restrict__ q,
                             const float* __restrict__ Cf,
                             const __bf16* __restrict__ nmw_pk,
                             const float* __restrict__ nmb,
                             float* __restrict__ out) {
    const int lane = threadIdx.x & 31;
    const int bt = blockIdx.x;
    const int ht = blockIdx.y;
    const int lm = lane & 15;
    const int g  = lane >> 4;
    const int arow = bt * 16 + lm;

    const float* srcs[3] = { pm + (size_t)arow * H_DIM,
                             Cf + (size_t)arow * H_DIM,
                             q  + (size_t)arow * H_DIM };
    v8f acc = {};
    #pragma unroll
    for (int segi = 0; segi < 3; ++segi) {
        const float* row = srcs[segi];
        for (int kt = 0; kt < 16; ++kt) {
            const int kb = kt * 32;
            v16bf afrag;
            #pragma unroll
            for (int ep = 0; ep < 8; ++ep) {
                int col = kb + a_pair_base(ep, g);
                afrag[2 * ep]     = (__bf16)row[col];
                afrag[2 * ep + 1] = (__bf16)row[col + 1];
            }
            const int ktg = segi * 16 + kt;                        // [0, 48)
            const v16bf bfrag =
                *(const v16bf*)(nmw_pk + (((size_t)ht * 48 + ktg) * 32 + lane) * 16);
            acc = __builtin_amdgcn_wmma_f32_16x16x32_bf16(
                false, afrag, false, bfrag, (short)0, acc, false, false);
        }
    }
    const int h = ht * 16 + lm;
    const float bias = nmb[h];
    #pragma unroll
    for (int i = 0; i < 8; ++i) {
        int bidx = bt * 16 + i + 8 * g;
        out[(size_t)bidx * H_DIM + h] = fmaxf(acc[i] + bias, 0.0f);
    }
}

// ---------------------------------------------------------------------
extern "C" void kernel_launch(void* const* d_in, const int* in_sizes, int n_in,
                              void* d_out, int out_size, void* d_ws, size_t ws_size,
                              hipStream_t stream) {
    const float* facts     = (const float*)d_in[0];
    const float* questions = (const float*)d_in[1];
    const float* prevM     = (const float*)d_in[2];
    const float* z1_w      = (const float*)d_in[3];
    const float* z1_b      = (const float*)d_in[4];
    const float* z2_w      = (const float*)d_in[5];
    // d_in[6] = z2_b: scalar shift, softmax-invariant -> unused
    const float* Wr        = (const float*)d_in[7];
    const float* br        = (const float*)d_in[8];
    const float* Ur        = (const float*)d_in[9];
    const float* bur       = (const float*)d_in[10];
    const float* W         = (const float*)d_in[11];
    const float* bw        = (const float*)d_in[12];
    const float* U         = (const float*)d_in[13];
    const float* bu        = (const float*)d_in[14];
    const float* nm_w      = (const float*)d_in[15];
    const float* nm_b      = (const float*)d_in[16];
    float* out = (float*)d_out;

    // workspace layout (bytes) -- total ~5.3 MB
    char* ws = (char*)d_ws;
    float*  logits  = (float*) (ws + 0);        // 32768 f32
    float*  G       = (float*) (ws + 131072);   // 32768 f32
    float*  rsWr    = (float*) (ws + 262144);   // 512 f32
    float*  rsW     = (float*) (ws + 264192);   // 512 f32
    float*  Cf      = (float*) (ws + 266240);   // 65536 f32 (C master copy)
    __bf16* Cbf0    = (__bf16*)(ws + 528384);   // 65536 bf16
    __bf16* Cbf1    = (__bf16*)(ws + 659456);   // 65536 bf16
    __bf16* z1w_pk  = (__bf16*)(ws + 790528);   // 2048*512 bf16 = 2 MB
    __bf16* Ur_pk   = (__bf16*)(ws + 2887680);  // 512*512 bf16 = 512 KB
    __bf16* U_pk    = (__bf16*)(ws + 3411968);  // 512*512 bf16 = 512 KB
    __bf16* nmw_pk  = (__bf16*)(ws + 3936256);  // 1536*512 bf16 = 1.5 MB

    init_kernel<<<256, 256, 0, stream>>>(logits, Cf, Cbf0);
    rowsum_kernel<<<1, 512, 0, stream>>>(Wr, W, rsWr, rsW);

    // one-time weight pre-swizzle into B-fragment register order
    pack_b_kernel<<<dim3(64, 32), 32, 0, stream>>>(z1_w, z1w_pk);
    pack_b_kernel<<<dim3(16, 32), 32, 0, stream>>>(Ur,   Ur_pk);
    pack_b_kernel<<<dim3(16, 32), 32, 0, stream>>>(U,    U_pk);
    pack_b_kernel<<<dim3(48, 32), 32, 0, stream>>>(nm_w, nmw_pk);

    // dominant GEMM: 32768 x 2048 x 512 in bf16 WMMA
    gate_gemm_kernel<<<dim3((B_DIM * S_DIM) / 16, H_DIM / 64), 128, 0, stream>>>(
        facts, questions, prevM, z1w_pk, z1_b, z2_w, logits);
    softmax_kernel<<<B_DIM, S_DIM, 0, stream>>>(logits, G);

    // sequential recurrence: 256 dependent step kernels, bf16 C ping-pong
    __bf16* cin = Cbf0;
    __bf16* cout = Cbf1;
    for (int s = 0; s < S_DIM; ++s) {
        scan_step_kernel<<<dim3(B_DIM / 16, H_DIM / 16), 32, 0, stream>>>(
            cin, cout, Cf, facts, G, Ur_pk, U_pk, rsWr, rsW, br, bur, bw, bu, s);
        __bf16* t = cin; cin = cout; cout = t;
    }

    final_kernel<<<dim3(B_DIM / 16, H_DIM / 16), 32, 0, stream>>>(
        prevM, questions, Cf, nmw_pk, nm_b, out);
}
